// SelfAttention_71571335020783
// MI455X (gfx1250) — matrix operand order repaired and checked by reading, not compiled, not run on hardware
//
#include <hip/hip_runtime.h>

// ---------------------------------------------------------------------------
// Self-attention (embed-dim attention variant) for MI455X / gfx1250.
// All 6 GEMMs run through v_wmma_f32_16x16x32_bf16 with f32 accumulation.
// Compute-bound (~2e11 FLOPs vs ~170MB traffic -> ~7us HBM at 23.3TB/s), so
// the matrix pipes must be saturated: bf16 (keeps f32 exponent range) at 8x
// the K-depth per issue of the f32 16x16x4 path.
//
// GEMM design: C = A(MxK,row) * B(KxN) with B supplied PRE-TRANSPOSED
// (Bt: NxK row-major) so A- and B-tile staging are identical coalesced
// global_load_b128 -> ds_store_b128 copies (no LDS scatter). LDS is
// double-buffered: tile t+1 is prefetched into registers while the 8 WMMAs
// of tile t execute; one barrier per K-step.
// ---------------------------------------------------------------------------

typedef unsigned short u16;
typedef __attribute__((ext_vector_type(8)))  u16    u16x8;
typedef __attribute__((ext_vector_type(16))) u16    u16x16;
typedef __attribute__((ext_vector_type(16))) __bf16 v16bf;
typedef __attribute__((ext_vector_type(8)))  float  v8f;

#define EMB   2048
#define SEQ   2048
#define BATCH 2

__device__ __forceinline__ u16 f32_to_bf16(float f) {
  unsigned u = __builtin_bit_cast(unsigned, f);
  unsigned r = u + 0x7FFFu + ((u >> 16) & 1u);   // round-to-nearest-even
  return (u16)(r >> 16);
}

// --------------------------- f32 -> bf16 convert ---------------------------
__global__ __launch_bounds__(256) void convert_bf16_kernel(
    const float* __restrict__ in, u16* __restrict__ out, int n) {
  int i = blockIdx.x * 256 + threadIdx.x;
  int stride = gridDim.x * 256;
  for (; i < n; i += stride) out[i] = f32_to_bf16(in[i]);
}

// ------------------- f32 -> bf16 convert + transpose (LDS tiled) -----------
// out[c][r] = bf16(in[r][c]);  R, C multiples of 32.  block (32,8)
__global__ __launch_bounds__(256) void convert_transpose_bf16_kernel(
    const float* __restrict__ in, u16* __restrict__ out, int R, int C) {
  __shared__ u16 tile[32][33];
  const int bx = blockIdx.x * 32;  // col base
  const int by = blockIdx.y * 32;  // row base
  const int tx = threadIdx.x, ty = threadIdx.y;
#pragma unroll
  for (int i = ty; i < 32; i += 8)
    tile[i][tx] = f32_to_bf16(in[(long)(by + i) * C + (bx + tx)]);
  __syncthreads();
#pragma unroll
  for (int i = ty; i < 32; i += 8)
    out[(long)(bx + i) * R + (by + tx)] = tile[tx][i];
}

// ------------------------------ WMMA GEMM ----------------------------------
// C[z] = A[z] (MxK row-major bf16) * Bt[z]^T  (Bt: NxK row-major bf16)
// OUT_BF16: store C as bf16 (else f32).  TRANS: store C transposed ([N][M]).
#define BM 128
#define BN 128
#define BK 32
#define LDT (BK + 8)   // +16B pad: keeps 16B alignment, staggers LDS banks

template <int OUT_BF16, int TRANS>
__global__ __launch_bounds__(256) void wmma_gemm_bf16(
    const u16* __restrict__ Ag, const u16* __restrict__ Btg, void* __restrict__ Cg,
    int M, int N, int K, long batchA, long batchB, long batchC) {
  __shared__ __align__(16) u16 As[2][BM][LDT];
  __shared__ __align__(16) u16 Bs[2][BN][LDT];   // Bs[n][k]

  const int tid   = threadIdx.x;
  const int lane  = tid & 31;
  const int wave  = tid >> 5;     // 0..7
  const int waveM = wave >> 1;    // 0..3 -> 32-row slice
  const int waveN = wave & 1;     // 0..1 -> 64-col slice
  const int l15   = lane & 15;
  const int lhi   = lane >> 4;

  const int blockRow = blockIdx.y * BM;
  const int blockCol = blockIdx.x * BN;

  // Per-thread staging slot: each thread copies 2 chunks of 8 halfs per tile
  // (rows crow and crow+64), identical pattern for A and Bt.
  const int crow = tid >> 2;          // 0..63
  const int ckc  = (tid & 3) * 8;     // 0,8,16,24
  const long rowSkip = 64L * K;

  const u16* aptr = Ag  + (long)blockIdx.z * batchA + (long)(blockRow + crow) * K + ckc;
  const u16* bptr = Btg + (long)blockIdx.z * batchB + (long)(blockCol + crow) * K + ckc;

  v8f acc[2][4];
#pragma unroll
  for (int r = 0; r < 2; ++r)
#pragma unroll
    for (int c = 0; c < 4; ++c) acc[r][c] = (v8f){0.f,0.f,0.f,0.f,0.f,0.f,0.f,0.f};

  // ---- preload tile 0 into buffer 0
  {
    u16x8 ra0 = *(const u16x8*)(aptr);
    u16x8 ra1 = *(const u16x8*)(aptr + rowSkip);
    u16x8 rb0 = *(const u16x8*)(bptr);
    u16x8 rb1 = *(const u16x8*)(bptr + rowSkip);
    aptr += BK; bptr += BK;
    *(u16x8*)(&As[0][crow][ckc])      = ra0;
    *(u16x8*)(&As[0][crow + 64][ckc]) = ra1;
    *(u16x8*)(&Bs[0][crow][ckc])      = rb0;
    *(u16x8*)(&Bs[0][crow + 64][ckc]) = rb1;
  }
  __syncthreads();

  const int NT = K / BK;
  for (int kt = 0; kt < NT - 1; ++kt) {
    const int p = kt & 1, q = p ^ 1;

    // prefetch tile kt+1 into registers (latency hidden behind WMMAs)
    u16x8 ra0 = *(const u16x8*)(aptr);
    u16x8 ra1 = *(const u16x8*)(aptr + rowSkip);
    u16x8 rb0 = *(const u16x8*)(bptr);
    u16x8 rb1 = *(const u16x8*)(bptr + rowSkip);
    aptr += BK; bptr += BK;

    // A fragment (ISA 16-bit A layout): lanes 0-15 hold K 0-7 & 16-23,
    // lanes 16-31 hold K 8-15 & 24-31; row M = lane&15.
    v16bf afrag[2], bfrag[4];
#pragma unroll
    for (int r = 0; r < 2; ++r) {
      const u16* s = &As[p][waveM * 32 + r * 16 + l15][0];
      u16x8 lo = *(const u16x8*)(s + lhi * 8);
      u16x8 hi = *(const u16x8*)(s + 16 + lhi * 8);
      afrag[r] = __builtin_bit_cast(v16bf, __builtin_shufflevector(
          lo, hi, 0,1,2,3,4,5,6,7,8,9,10,11,12,13,14,15));
    }
    // B fragment: lanes 0-15 hold K 0-7 & 16-23, lanes 16-31 hold K 8-15 & 24-31; col = lane&15.
#pragma unroll
    for (int c = 0; c < 4; ++c) {
      const u16* s = &Bs[p][waveN * 64 + c * 16 + l15][0];
      u16x8 lo = *(const u16x8*)(s + lhi * 8);
      u16x8 hi = *(const u16x8*)(s + 16 + lhi * 8);
      bfrag[c] = __builtin_bit_cast(v16bf, __builtin_shufflevector(
          lo, hi, 0,1,2,3,4,5,6,7,8,9,10,11,12,13,14,15));
    }

#pragma unroll
    for (int r = 0; r < 2; ++r)
#pragma unroll
      for (int c = 0; c < 4; ++c)
        acc[r][c] = __builtin_amdgcn_wmma_f32_16x16x32_bf16(
            false, afrag[r], false, bfrag[c], (short)0, acc[r][c], false, false);

    // commit prefetched tile into the other buffer
    *(u16x8*)(&As[q][crow][ckc])      = ra0;
    *(u16x8*)(&As[q][crow + 64][ckc]) = ra1;
    *(u16x8*)(&Bs[q][crow][ckc])      = rb0;
    *(u16x8*)(&Bs[q][crow + 64][ckc]) = rb1;
    __syncthreads();
  }

  // ---- epilogue K-step (buffer (NT-1)&1)
  {
    const int p = (NT - 1) & 1;
    v16bf afrag[2], bfrag[4];
#pragma unroll
    for (int r = 0; r < 2; ++r) {
      const u16* s = &As[p][waveM * 32 + r * 16 + l15][0];
      u16x8 lo = *(const u16x8*)(s + lhi * 8);
      u16x8 hi = *(const u16x8*)(s + 16 + lhi * 8);
      afrag[r] = __builtin_bit_cast(v16bf, __builtin_shufflevector(
          lo, hi, 0,1,2,3,4,5,6,7,8,9,10,11,12,13,14,15));
    }
#pragma unroll
    for (int c = 0; c < 4; ++c) {
      const u16* s = &Bs[p][waveN * 64 + c * 16 + l15][0];
      u16x8 lo = *(const u16x8*)(s + lhi * 8);
      u16x8 hi = *(const u16x8*)(s + 16 + lhi * 8);
      bfrag[c] = __builtin_bit_cast(v16bf, __builtin_shufflevector(
          lo, hi, 0,1,2,3,4,5,6,7,8,9,10,11,12,13,14,15));
    }
#pragma unroll
    for (int r = 0; r < 2; ++r)
#pragma unroll
      for (int c = 0; c < 4; ++c)
        acc[r][c] = __builtin_amdgcn_wmma_f32_16x16x32_bf16(
            false, afrag[r], false, bfrag[c], (short)0, acc[r][c], false, false);
  }

  // C/D layout: VGPR i -> M = i (lanes 0-15) or 8+i (lanes 16-31); N = lane&15.
  const long cb = (long)blockIdx.z * batchC;
#pragma unroll
  for (int r = 0; r < 2; ++r) {
#pragma unroll
    for (int c = 0; c < 4; ++c) {
      const int tm = blockRow + waveM * 32 + r * 16;
      const int tn = blockCol + waveN * 64 + c * 16;
#pragma unroll
      for (int i = 0; i < 8; ++i) {
        int row = tm + lhi * 8 + i;
        int col = tn + l15;
        float val = acc[r][c][i];
        long idx = TRANS ? ((long)col * M + row) : ((long)row * N + col);
        if (OUT_BF16) ((u16*)Cg)[cb + idx] = f32_to_bf16(val);
        else          ((float*)Cg)[cb + idx] = val;
      }
    }
  }
}

// ------------------------- softmax (scale folded in) -----------------------
// One 256-thread block per row of EMB f32 scores; emits bf16 weights.
__global__ __launch_bounds__(256) void softmax_scale_bf16(
    const float* __restrict__ scores, u16* __restrict__ w, float scale) {
  const long row = blockIdx.x;
  const float* in = scores + row * (long)EMB;
  u16* out = w + row * (long)EMB;
  __shared__ float red[256];
  const int tid = threadIdx.x;

  float vals[EMB / 256];
  float m = -3.4e38f;
#pragma unroll
  for (int j = 0; j < EMB / 256; ++j) {
    vals[j] = in[tid + j * 256] * scale;
    m = fmaxf(m, vals[j]);
  }
  red[tid] = m;  __syncthreads();
#pragma unroll
  for (int s = 128; s > 0; s >>= 1) {
    if (tid < s) red[tid] = fmaxf(red[tid], red[tid + s]);
    __syncthreads();
  }
  m = red[0];  __syncthreads();

  float sum = 0.f;
#pragma unroll
  for (int j = 0; j < EMB / 256; ++j) {
    vals[j] = __expf(vals[j] - m);
    sum += vals[j];
  }
  red[tid] = sum;  __syncthreads();
#pragma unroll
  for (int s = 128; s > 0; s >>= 1) {
    if (tid < s) red[tid] += red[tid + s];
    __syncthreads();
  }
  const float inv = 1.0f / red[0];
#pragma unroll
  for (int j = 0; j < EMB / 256; ++j)
    out[tid + j * 256] = f32_to_bf16(vals[j] * inv);
}

// ------------------------------- launch ------------------------------------
extern "C" void kernel_launch(void* const* d_in, const int* in_sizes, int n_in,
                              void* d_out, int out_size, void* d_ws, size_t ws_size,
                              hipStream_t stream) {
  const float* x  = (const float*)d_in[0];
  const float* Wq = (const float*)d_in[1];
  const float* Wk = (const float*)d_in[2];
  const float* Wv = (const float*)d_in[3];
  const float* Wo = (const float*)d_in[4];
  float* out = (float*)d_out;

  const size_t EE  = (size_t)EMB * EMB;
  const size_t BSE = (size_t)BATCH * SEQ * EMB;

  char* ws = (char*)d_ws;
  size_t off = 0;
  u16*   xb     = (u16*)(ws + off); off += BSE * 2;
  u16*   Wqt    = (u16*)(ws + off); off += EE * 2;   // transposed bf16 weights
  u16*   Wkt    = (u16*)(ws + off); off += EE * 2;
  u16*   Wvt    = (u16*)(ws + off); off += EE * 2;
  u16*   Wot    = (u16*)(ws + off); off += EE * 2;
  u16*   q_t    = (u16*)(ws + off); off += BSE * 2;  // [b][e][s]
  u16*   k_t    = (u16*)(ws + off); off += BSE * 2;  // [b][e][s]
  u16*   vmat   = (u16*)(ws + off); off += BSE * 2;  // [b][s][e]
  float* scores = (float*)(ws + off); off += (size_t)BATCH * EE * 4; // [b][e][e]
  u16*   weights = q_t;   // reuse: q_t dead after scores GEMM
  u16*   attn_t  = k_t;   // reuse: k_t dead after scores GEMM; [b][s][e]

  // conversions: x plain; weights transposed (so GEMM B-operand is NxK)
  convert_bf16_kernel<<<4096, 256, 0, stream>>>(x, xb, (int)BSE);
  dim3 tb(32, 8), tg(EMB / 32, EMB / 32);
  convert_transpose_bf16_kernel<<<tg, tb, 0, stream>>>(Wq, Wqt, EMB, EMB);
  convert_transpose_bf16_kernel<<<tg, tb, 0, stream>>>(Wk, Wkt, EMB, EMB);
  convert_transpose_bf16_kernel<<<tg, tb, 0, stream>>>(Wv, Wvt, EMB, EMB);
  convert_transpose_bf16_kernel<<<tg, tb, 0, stream>>>(Wo, Wot, EMB, EMB);

  dim3 g(EMB / BN, SEQ / BM, BATCH);   // 16 x 16 x 2
  const long SE  = (long)SEQ * EMB;
  const long EEl = (long)EMB * EMB;

  // q_t[b] = (x[b] Wq)^T   [e][s] bf16      (A=x, Bt=Wq^T)
  wmma_gemm_bf16<1, 1><<<g, 256, 0, stream>>>(xb, Wqt, q_t, SEQ, EMB, EMB, SE, 0, SE);
  // k_t[b] = (x[b] Wk)^T   [e][s] bf16
  wmma_gemm_bf16<1, 1><<<g, 256, 0, stream>>>(xb, Wkt, k_t, SEQ, EMB, EMB, SE, 0, SE);
  // v[b]   =  x[b] Wv      [s][e] bf16      (v IS the pre-transposed B for attn GEMM)
  wmma_gemm_bf16<1, 0><<<g, 256, 0, stream>>>(xb, Wvt, vmat, SEQ, EMB, EMB, SE, 0, SE);
  // scores[b] = q_t[b] k_t[b]^T   [e][e] f32   (contraction over s; Bt = k_t)
  wmma_gemm_bf16<0, 0><<<g, 256, 0, stream>>>(q_t, k_t, scores, EMB, EMB, SEQ, SE, SE, EEl);
  // weights[b] = softmax(scores[b] / sqrt(64))   bf16
  softmax_scale_bf16<<<BATCH * EMB, 256, 0, stream>>>(scores, weights, 0.125f);
  // attn_t[b] = (weights[b] v[b]^T)^T = [s][e] bf16   (A=weights ExE, Bt=v SxE)
  wmma_gemm_bf16<1, 1><<<g, 256, 0, stream>>>(weights, vmat, attn_t, EMB, SEQ, EMB, EEl, SE, SE);
  // out[b] = attn_t[b] Wo   [s][e] f32 -> d_out   (Bt = Wo^T)
  wmma_gemm_bf16<0, 0><<<g, 256, 0, stream>>>(attn_t, Wot, out, SEQ, EMB, EMB, SE, 0, SE);
}
